// MultiSimilarityLoss_81698867904771
// MI455X (gfx1250) — matrix operand order repaired and checked by reading, not compiled, not run on hardware
//
#include <hip/hip_runtime.h>
#include <hip/hip_bf16.h>
#include <math.h>

// ---------------------------------------------------------------------------
// MultiSimilarityLoss, fused: sim = F F^T via v_wmma_f32_16x16x32_bf16,
// never materialized. B=8192, D=128. Branchless cndmask epilogues,
// deep-clause B loads + interleaved dual accumulation chains,
// 4-way column split for occupancy (2048 waves).
// ---------------------------------------------------------------------------

typedef __attribute__((ext_vector_type(16))) __bf16 v16bf;
typedef __attribute__((ext_vector_type(8)))  __bf16 v8bf;
typedef __attribute__((ext_vector_type(8)))  float  v8f;

#define NB 8192
#define ND 128
#define NPART 4
#define ONE_EPS 0.99999f
#define MG 0.1f
// exp(a*(s-0.5)) == exp2(cf*s + of)
#define CFP (-2.885390082f)    // -2  * log2(e)
#define OFP ( 1.442695041f)    // +2  * log2(e) * 0.5
#define CFN ( 57.70780163f)    // +40 * log2(e)
#define OFN (-28.85390082f)    // -40 * log2(e) * 0.5

#define WMMA_BF16(A, B, C) \
    __builtin_amdgcn_wmma_f32_16x16x32_bf16(false, (A), false, (B), (short)0, (C), false, false)

// ---- fp32 -> bf16 ---------------------------------------------------------
__global__ __launch_bounds__(256) void msl_convert(const float* __restrict__ f,
                                                   __bf16* __restrict__ o, int n) {
    int i = blockIdx.x * blockDim.x + threadIdx.x;
    if (i < n) o[i] = (__bf16)f[i];
}

// ---- WMMA A-fragment loader (contiguous k = 16*hf + s, wave32) -------------
__device__ inline v16bf load_frag_a(const __bf16* F, int rb, int k0, int lane) {
    int lr = lane & 15, hf = lane >> 4;
    unsigned off = (unsigned)(rb + lr) * ND + (unsigned)(k0 + hf * 16);
    v8bf lo = *(const v8bf*)(F + off);
    v8bf hi = *(const v8bf*)(F + off + 8);
    v16bf a;
#pragma unroll
    for (int i = 0; i < 8; ++i) { a[i] = lo[i]; a[i + 8] = hi[i]; }
    return a;
}

// Two adjacent 16x16 sim tiles (cols cb, cb+16), K=128. All B loads issued
// up front (consumption order), two interleaved WMMA chains. Diagonal lane
// poked to 2.0 on diagonal tiles (fails s < 1-eps -> never a positive).
__device__ inline void sim_tile2(const __bf16* __restrict__ F, const v16bf af[4],
                                 int cb, int rb, int lr, int hf, int dsel,
                                 v8f& ca, v8f& cc) {
    unsigned ra = (unsigned)(cb + lr) * ND + (unsigned)(hf * 16);
    unsigned rc = ra + 16u * ND;
    v16bf b0 = *(const v16bf*)(F + ra);
    v16bf d0 = *(const v16bf*)(F + rc);
    v16bf b1 = *(const v16bf*)(F + ra + 32);
    v16bf d1 = *(const v16bf*)(F + rc + 32);
    v16bf b2 = *(const v16bf*)(F + ra + 64);
    v16bf d2 = *(const v16bf*)(F + rc + 64);
    v16bf b3 = *(const v16bf*)(F + ra + 96);
    v16bf d3 = *(const v16bf*)(F + rc + 96);
    v8f x = {}, y = {};
    x = WMMA_BF16(af[0], b0, x);
    y = WMMA_BF16(af[0], d0, y);
    x = WMMA_BF16(af[1], b1, x);
    y = WMMA_BF16(af[1], d1, y);
    x = WMMA_BF16(af[2], b2, x);
    y = WMMA_BF16(af[2], d2, y);
    x = WMMA_BF16(af[3], b3, x);
    y = WMMA_BF16(af[3], d3, y);
    if (cb == rb) {
#pragma unroll
        for (int v = 0; v < 8; ++v) x[v] = (dsel == v) ? 2.0f : x[v];
    }
    if (cb + 16 == rb) {
#pragma unroll
        for (int v = 0; v < 8; ++v) y[v] = (dsel == v) ? 2.0f : y[v];
    }
    ca = x; cc = y;
}

// ---- Pass A: per-row min(pos) / max(neg), split over column quarters ------
__global__ __launch_bounds__(256) void msl_minmax(const __bf16* __restrict__ F,
                                                  const int* __restrict__ labels,
                                                  float* __restrict__ minp_part,
                                                  float* __restrict__ maxn_part) {
    __shared__ int slab[NB];
    for (int i = threadIdx.x; i < NB; i += blockDim.x) slab[i] = labels[i];
    __syncthreads();

    int lane = threadIdx.x & 31;
    int wid  = blockIdx.x * (blockDim.x >> 5) + (threadIdx.x >> 5); // 0..2047
    int rb   = (wid >> 2) * 16;
    int part = wid & (NPART - 1);
    int c0   = part * (NB / NPART), c1 = c0 + NB / NPART;
    int lr = lane & 15, hf = lane >> 4;
    int dsel = lr - hf * 8;

    v16bf afrag[4];
#pragma unroll
    for (int kk = 0; kk < 4; ++kk) afrag[kk] = load_frag_a(F, rb, kk * 32, lane);

    int rlab[8];
#pragma unroll
    for (int v = 0; v < 8; ++v) rlab[v] = slab[rb + hf * 8 + v];

    float mnp[8], mxn[8];
#pragma unroll
    for (int v = 0; v < 8; ++v) { mnp[v] = __builtin_inff(); mxn[v] = -__builtin_inff(); }

    for (int cb = c0; cb < c1; cb += 32) {
        int jla = slab[cb + lr];          // issue LDS reads first
        int jlb = slab[cb + 16 + lr];
        v8f ca, cc;
        sim_tile2(F, afrag, cb, rb, lr, hf, dsel, ca, cc);
#pragma unroll
        for (int v = 0; v < 8; ++v) {
            float s  = ca[v];
            bool same = (rlab[v] == jla);
            float sp = (same & (s < ONE_EPS)) ? s : __builtin_inff();
            float sn = same ? -__builtin_inff() : s;
            mnp[v] = (sp < mnp[v]) ? sp : mnp[v];
            mxn[v] = (sn > mxn[v]) ? sn : mxn[v];
        }
#pragma unroll
        for (int v = 0; v < 8; ++v) {
            float s  = cc[v];
            bool same = (rlab[v] == jlb);
            float sp = (same & (s < ONE_EPS)) ? s : __builtin_inff();
            float sn = same ? -__builtin_inff() : s;
            mnp[v] = (sp < mnp[v]) ? sp : mnp[v];
            mxn[v] = (sn > mxn[v]) ? sn : mxn[v];
        }
    }
#pragma unroll
    for (int v = 0; v < 8; ++v) {
#pragma unroll
        for (int m = 1; m <= 8; m <<= 1) {
            float a = __shfl_xor(mnp[v], m, 32);
            float b = __shfl_xor(mxn[v], m, 32);
            mnp[v] = (a < mnp[v]) ? a : mnp[v];
            mxn[v] = (b > mxn[v]) ? b : mxn[v];
        }
    }
    if (lr == 0) {
#pragma unroll
        for (int v = 0; v < 8; ++v) {
            int r = rb + hf * 8 + v;
            minp_part[part * NB + r] = mnp[v];
            maxn_part[part * NB + r] = mxn[v];
        }
    }
}

// ---- Pass B: masked exp sums (single exp2 per element) --------------------
__global__ __launch_bounds__(256) void msl_loss(const __bf16* __restrict__ F,
                                                const int* __restrict__ labels,
                                                const float* __restrict__ minp_part,
                                                const float* __restrict__ maxn_part,
                                                float* __restrict__ psum_part,
                                                float* __restrict__ nsum_part) {
    __shared__ int slab[NB];
    for (int i = threadIdx.x; i < NB; i += blockDim.x) slab[i] = labels[i];
    __syncthreads();

    int lane = threadIdx.x & 31;
    int wid  = blockIdx.x * (blockDim.x >> 5) + (threadIdx.x >> 5);
    int rb   = (wid >> 2) * 16;
    int part = wid & (NPART - 1);
    int c0   = part * (NB / NPART), c1 = c0 + NB / NPART;
    int lr = lane & 15, hf = lane >> 4;
    int dsel = lr - hf * 8;

    v16bf afrag[4];
#pragma unroll
    for (int kk = 0; kk < 4; ++kk) afrag[kk] = load_frag_a(F, rb, kk * 32, lane);

    int rlab[8]; float pthr[8], nthrN[8], psum[8], nsum[8];
#pragma unroll
    for (int v = 0; v < 8; ++v) {
        int r = rb + hf * 8 + v;
        rlab[v] = slab[r];
        float mn = minp_part[r];
        float mx = maxn_part[r];
#pragma unroll
        for (int p = 1; p < NPART; ++p) {
            float a = minp_part[p * NB + r];
            float b = maxn_part[p * NB + r];
            mn = (a < mn) ? a : mn;                      // combined min_pos
            mx = (b > mx) ? b : mx;                      // combined max_neg
        }
        float pt = mx + MG;
        pthr[v]  = (pt < ONE_EPS) ? pt : ONE_EPS;        // pos: s < pthr
        nthrN[v] = MG - mn;                              // neg: -s < MG-min_pos
        psum[v] = 0.0f; nsum[v] = 0.0f;
    }

    for (int cb = c0; cb < c1; cb += 32) {
        int jla = slab[cb + lr];
        int jlb = slab[cb + 16 + lr];
        v8f ca, cc;
        sim_tile2(F, afrag, cb, rb, lr, hf, dsel, ca, cc);
#pragma unroll
        for (int v = 0; v < 8; ++v) {
            float s  = ca[v];
            bool same = (rlab[v] == jla);
            float u  = same ? s : -s;
            float t  = same ? pthr[v] : nthrN[v];
            float cf = same ? CFP : CFN;
            float of = same ? OFP : OFN;
            float yy = (u < t) ? __builtin_fmaf(cf, s, of) : -200.0f;
            float e  = __builtin_amdgcn_exp2f(yy);       // exp2(-200) == 0
            float pe = same ? e : 0.0f;
            psum[v] += pe;
            nsum[v] += e - pe;
        }
#pragma unroll
        for (int v = 0; v < 8; ++v) {
            float s  = cc[v];
            bool same = (rlab[v] == jlb);
            float u  = same ? s : -s;
            float t  = same ? pthr[v] : nthrN[v];
            float cf = same ? CFP : CFN;
            float of = same ? OFP : OFN;
            float yy = (u < t) ? __builtin_fmaf(cf, s, of) : -200.0f;
            float e  = __builtin_amdgcn_exp2f(yy);
            float pe = same ? e : 0.0f;
            psum[v] += pe;
            nsum[v] += e - pe;
        }
    }
#pragma unroll
    for (int v = 0; v < 8; ++v) {
#pragma unroll
        for (int m = 1; m <= 8; m <<= 1) {
            psum[v] += __shfl_xor(psum[v], m, 32);
            nsum[v] += __shfl_xor(nsum[v], m, 32);
        }
    }
    if (lr == 0) {
#pragma unroll
        for (int v = 0; v < 8; ++v) {
            int r = rb + hf * 8 + v;
            psum_part[part * NB + r] = psum[v];
            nsum_part[part * NB + r] = nsum[v];
        }
    }
}

// ---- deterministic combine + row loss + reduction -------------------------
__global__ __launch_bounds__(256) void msl_final(const float* __restrict__ mnp,
                                                 const float* __restrict__ mxn,
                                                 const float* __restrict__ ps,
                                                 const float* __restrict__ ns,
                                                 float* __restrict__ out) {
    __shared__ float sd[256];
    float acc = 0.0f;
    for (int r = threadIdx.x; r < NB; r += 256) {
        float mn = mnp[r];
        float mx = mxn[r];
        float p  = ps[r];
        float n  = ns[r];
#pragma unroll
        for (int q = 1; q < NPART; ++q) {
            float a = mnp[q * NB + r];
            float b = mxn[q * NB + r];
            mn = (a < mn) ? a : mn;
            mx = (b > mx) ? b : mx;
            p += ps[q * NB + r];
            n += ns[q * NB + r];
        }
        bool valid = (mn < 3.0e38f) & (mx > -3.0e38f) & (p > 0.0f) & (n > 0.0f);
        float l = 0.5f * log1pf(p) + 0.025f * log1pf(n);
        acc += valid ? l : 0.0f;
    }
    sd[threadIdx.x] = acc;
    __syncthreads();
    for (int k = 128; k > 0; k >>= 1) {
        if (threadIdx.x < k) sd[threadIdx.x] += sd[threadIdx.x + k];
        __syncthreads();
    }
    if (threadIdx.x == 0) out[0] = sd[0];
}

extern "C" void kernel_launch(void* const* d_in, const int* in_sizes, int n_in,
                              void* d_out, int out_size, void* d_ws, size_t ws_size,
                              hipStream_t stream) {
    const float* feat   = (const float*)d_in[0];   // [8192,128] f32
    const int*   labels = (const int*)d_in[1];     // [8192,1] i32
    float*       out    = (float*)d_out;           // scalar f32

    char* w = (char*)d_ws;
    __bf16* Fb   = (__bf16*)w;                         // 2 MB
    float*  minp = (float*)(w + (size_t)NB * ND * 2);  // [NPART][NB]
    float*  maxn = minp + NPART * NB;                  // [NPART][NB]
    float*  psm  = maxn + NPART * NB;                  // [NPART][NB]
    float*  nsm  = psm  + NPART * NB;                  // [NPART][NB]

    msl_convert<<<(NB * ND) / 256, 256, 0, stream>>>(feat, Fb, NB * ND);
    msl_minmax <<<256, 256, 0, stream>>>(Fb, labels, minp, maxn);
    msl_loss   <<<256, 256, 0, stream>>>(Fb, labels, minp, maxn, psm, nsm);
    msl_final  <<<1, 256, 0, stream>>>(minp, maxn, psm, nsm, out);
}
